// HumanLikeWorkingMemory_66529043415105
// MI455X (gfx1250) — matrix operand order repaired and checked by reading, not compiled, not run on hardware
//
#include <hip/hip_runtime.h>
#include <stdint.h>

#define DM 1024
#define BATCH_N 16384
#define CAP 7

typedef __attribute__((ext_vector_type(16))) __bf16 v16bf;
typedef __attribute__((ext_vector_type(8)))  float  v8f;

union FragBF { unsigned int u[8]; v16bf v; };

__device__ __forceinline__ unsigned int bf16_rne(float f) {
    unsigned int u = __float_as_uint(f);
    u += 0x7FFFu + ((u >> 16) & 1u);
    return u >> 16;
}
__device__ __forceinline__ float bf16_f32(unsigned int h) {
    return __uint_as_float(h << 16);
}

// ---------------------------------------------------------------------------
// K0: transpose+split Wg (f32 [k][n]) -> Whi_t/Wlo_t (bf16 [n][k])
// ---------------------------------------------------------------------------
__global__ __launch_bounds__(256) void wm_prep_w(
    const float* __restrict__ Wg,
    unsigned short* __restrict__ Whi, unsigned short* __restrict__ Wlo)
{
    int idx = blockIdx.x * 256 + threadIdx.x;      // over 1024*1024, k-major
    int k = idx >> 10;
    int n = idx & (DM - 1);
    float w = Wg[idx];
    unsigned int h = bf16_rne(w);
    unsigned int l = bf16_rne(w - bf16_f32(h));
    Whi[(size_t)n * DM + k] = (unsigned short)h;
    Wlo[(size_t)n * DM + k] = (unsigned short)l;
}

// ---------------------------------------------------------------------------
// K1: imp[b] = sigmoid(dot(X[b], Wi) + bi)   (one wave per row)
// ---------------------------------------------------------------------------
__global__ __launch_bounds__(256) void wm_importance(
    const float* __restrict__ X, const float* __restrict__ Wi,
    const float* __restrict__ bi, float* __restrict__ imp)
{
    int gid  = blockIdx.x * 256 + threadIdx.x;
    int row  = gid >> 5;
    int lane = threadIdx.x & 31;
    const float* x = X + (size_t)row * DM;
    float s = 0.f;
#pragma unroll
    for (int k0 = 0; k0 < DM; k0 += 128) {
        float4 a = *(const float4*)(x  + k0 + lane * 4);
        float4 w = *(const float4*)(Wi + k0 + lane * 4);
        s += a.x * w.x + a.y * w.y + a.z * w.z + a.w * w.w;
    }
#pragma unroll
    for (int m = 16; m >= 1; m >>= 1) s += __shfl_xor(s, m, 32);
    if (lane == 0) imp[row] = 1.f / (1.f + __expf(-(s + bi[0])));
}

// ---------------------------------------------------------------------------
// K2a: minimal serial state recurrence. Stores per-step post-update state:
//   impS[b*8+c] = importance of slot c after step b
//   srcS[b*8+c] = batch row whose update vector occupies slot c after step b
// Occupancy is implicit: slot c occupied at step b iff b >= c.
// ---------------------------------------------------------------------------
__device__ __forceinline__ void scan_store(int b, const float imp[CAP], const int src[CAP],
                                           float* __restrict__ impS, int* __restrict__ srcS)
{
    float4 f0 = make_float4(imp[0], imp[1], imp[2], imp[3]);
    float4 f1 = make_float4(imp[4], imp[5], imp[6], 0.f);
    int4   i0 = make_int4(src[0], src[1], src[2], src[3]);
    int4   i1 = make_int4(src[4], src[5], src[6], 0);
    *(float4*)(impS + (size_t)b * 8)     = f0;
    *(float4*)(impS + (size_t)b * 8 + 4) = f1;
    *(int4*)  (srcS + (size_t)b * 8)     = i0;
    *(int4*)  (srcS + (size_t)b * 8 + 4) = i1;
}

__device__ __forceinline__ void scan_step(float ix, int b,
                                          float imp[CAP], float age[CAP], int src[CAP],
                                          float* __restrict__ impS, int* __restrict__ srcS)
{
    // argmin (first-min tie-break, matches jnp.argmin)
    float vmin = imp[0]; int jmin = 0;
#pragma unroll
    for (int c = 1; c < CAP; ++c) if (imp[c] < vmin) { vmin = imp[c]; jmin = c; }
    bool  doW  = ix > 0.1f;
    float nimp = fmaxf(0.1f, ix);
#pragma unroll
    for (int c = 0; c < CAP; ++c) {
        bool  wm = doW && (jmin == c);
        float na = wm ? 1.f : age[c] + 1.f;          // write resets to 0, then +1
        float ni = wm ? nimp : ((na > 50.f) ? imp[c] * 0.9f : imp[c]);
        src[c] = wm ? b : src[c];
        age[c] = na;
        imp[c] = ni;
    }
    scan_store(b, imp, src, impS, srcS);
}

__global__ void wm_scan_state(const float* __restrict__ impx,
                              float* __restrict__ impS, int* __restrict__ srcS)
{
    if (threadIdx.x != 0 || blockIdx.x != 0) return;
    float imp[CAP], age[CAP]; int src[CAP];
#pragma unroll
    for (int c = 0; c < CAP; ++c) { imp[c] = 0.f; age[c] = 0.f; src[c] = 0; }

    // warm-up: step b always writes (first empty) slot b
#pragma unroll
    for (int b = 0; b < CAP; ++b) {
        float ix = impx[b];
        imp[b] = fmaxf(0.1f, ix);
        age[b] = 0.f;
        src[b] = b;
#pragma unroll
        for (int c = 0; c < CAP; ++c) age[c] += 1.f;   // ages <= 7: no decay possible
        scan_store(b, imp, src, impS, srcS);
    }
    // b = 7 (odd one out so the main loop starts 16B-aligned)
    scan_step(impx[CAP], CAP, imp, age, src, impS, srcS);
    // main loop: no occupancy logic; float4-batched importance loads
    for (int b0 = 8; b0 < BATCH_N; b0 += 4) {
        float4 q = *(const float4*)(impx + b0);
        scan_step(q.x, b0,     imp, age, src, impS, srcS);
        scan_step(q.y, b0 + 1, imp, age, src, impS, srcS);
        scan_step(q.z, b0 + 2, imp, age, src, impS, srcS);
        scan_step(q.w, b0 + 3, imp, age, src, impS, srcS);
    }
}

// ---------------------------------------------------------------------------
// K2b: parallel per-step softmax weights from stored state (1 thread / step)
// ---------------------------------------------------------------------------
__global__ __launch_bounds__(256) void wm_weights(
    const float* __restrict__ impS, float* __restrict__ wv)
{
    int b = blockIdx.x * 256 + threadIdx.x;
    float4 f0 = *(const float4*)(impS + (size_t)b * 8);
    float4 f1 = *(const float4*)(impS + (size_t)b * 8 + 4);
    float im[CAP] = { f0.x, f0.y, f0.z, f0.w, f1.x, f1.y, f1.z };

    float wsum = 0.f, mx = -3.0e38f;
#pragma unroll
    for (int c = 0; c < CAP; ++c) {
        bool occ = (b >= c);
        if (occ) { wsum += im[c]; mx = fmaxf(mx, im[c]); }
    }
    float e[CAP], es = 0.f;
#pragma unroll
    for (int c = 0; c < CAP; ++c) {
        e[c] = (b >= c) ? __expf(im[c] - mx) : 0.f;
        es += e[c];
    }
    float inv = (es > 0.f) ? 1.f / es : 0.f;
    bool fb = !(wsum > 0.f);                 // fallback: slot 0 is always occupied
    float w[CAP];
#pragma unroll
    for (int c = 0; c < CAP; ++c)
        w[c] = fb ? ((c == 0) ? 1.f : 0.f) : e[c] * inv;

    float4 w0 = make_float4(w[0], w[1], w[2], w[3]);
    float4 w1 = make_float4(w[4], w[5], w[6], 0.f);
    *(float4*)(wv + (size_t)b * 8)     = w0;
    *(float4*)(wv + (size_t)b * 8 + 4) = w1;
}

// ---------------------------------------------------------------------------
// K3: U = X @ Wg + bg via split-bf16 WMMA (hi*hi + lo*hi + hi*lo).
// Block: 8 waves, 128(M) x 64(N); each wave: 16(M) x 64(N), K-loop step 32.
// ---------------------------------------------------------------------------
__global__ __launch_bounds__(256) void wm_gemm(
    const float* __restrict__ X,
    const unsigned short* __restrict__ Whi, const unsigned short* __restrict__ Wlo,
    const float* __restrict__ bg, float* __restrict__ U)
{
    const int lane = threadIdx.x & 31;
    const int wave = threadIdx.x >> 5;
    const int half = lane >> 4;          // 0 or 1
    const int l16  = lane & 15;
    const int M0 = blockIdx.y * 128 + wave * 16;
    const int N0 = blockIdx.x * 64;

    v8f acc[4] = {};

    const float* xrow = X + (size_t)(M0 + l16) * DM;

    for (int kk = 0; kk < DM; kk += 32) {
        // ---- A fragment: 16x32 bf16, contiguous layout (lane<16: K {0..15}; lane>=16: K {16..31})
        const int kA = kk + half * 16;
        float4 f0 = *(const float4*)(xrow + kA);
        float4 f1 = *(const float4*)(xrow + kA + 4);
        float4 f2 = *(const float4*)(xrow + kA + 8);
        float4 f3 = *(const float4*)(xrow + kA + 12);
        float f[16] = { f0.x, f0.y, f0.z, f0.w, f1.x, f1.y, f1.z, f1.w,
                        f2.x, f2.y, f2.z, f2.w, f3.x, f3.y, f3.z, f3.w };
        FragBF ahi, alo;
#pragma unroll
        for (int v = 0; v < 8; ++v) {
            unsigned int h0 = bf16_rne(f[2 * v]);
            unsigned int h1 = bf16_rne(f[2 * v + 1]);
            ahi.u[v] = h0 | (h1 << 16);
            unsigned int r0 = bf16_rne(f[2 * v]     - bf16_f32(h0));
            unsigned int r1 = bf16_rne(f[2 * v + 1] - bf16_f32(h1));
            alo.u[v] = r0 | (r1 << 16);
        }

        // ---- B fragments from transposed bf16 W: lane<16 K {0..15}, lane>=16 K {16..31}
        const int kB = kk + half * 16;
#pragma unroll
        for (int t = 0; t < 4; ++t) {
            const size_t boff = (size_t)(N0 + t * 16 + l16) * DM + kB;
            const uint4* ph = (const uint4*)(Whi + boff);
            const uint4* pl = (const uint4*)(Wlo + boff);
            uint4 h0 = ph[0], h1 = ph[1];
            uint4 l0 = pl[0], l1 = pl[1];
            FragBF bhi, blo;
            bhi.u[0] = h0.x; bhi.u[1] = h0.y; bhi.u[2] = h0.z; bhi.u[3] = h0.w;
            bhi.u[4] = h1.x; bhi.u[5] = h1.y; bhi.u[6] = h1.z; bhi.u[7] = h1.w;
            blo.u[0] = l0.x; blo.u[1] = l0.y; blo.u[2] = l0.z; blo.u[3] = l0.w;
            blo.u[4] = l1.x; blo.u[5] = l1.y; blo.u[6] = l1.z; blo.u[7] = l1.w;

            acc[t] = __builtin_amdgcn_wmma_f32_16x16x32_bf16(
                false, ahi.v, false, bhi.v, (short)0, acc[t], false, false);
            acc[t] = __builtin_amdgcn_wmma_f32_16x16x32_bf16(
                false, alo.v, false, bhi.v, (short)0, acc[t], false, false);
            acc[t] = __builtin_amdgcn_wmma_f32_16x16x32_bf16(
                false, ahi.v, false, blo.v, (short)0, acc[t], false, false);
        }
    }

    // ---- epilogue: C/D layout (VGPR r: lanes0-15 M=r, lanes16-31 M=r+8; N=lane%16)
#pragma unroll
    for (int t = 0; t < 4; ++t) {
        const int col = N0 + t * 16 + l16;
        const float bias = bg[col];
#pragma unroll
        for (int r = 0; r < 8; ++r) {
            const int row = M0 + half * 8 + r;
            U[(size_t)row * DM + col] = acc[t][r] + bias;
        }
    }
}

// ---------------------------------------------------------------------------
// K4: out[b] = sum_c w[b][c] * U[src[b][c]]   (one block per batch row)
// ---------------------------------------------------------------------------
__global__ __launch_bounds__(256) void wm_combine(
    const float* __restrict__ U, const float* __restrict__ wv,
    const int* __restrict__ wid, float* __restrict__ out)
{
    const int b = blockIdx.x;
    const float* w8 = wv  + (size_t)b * 8;
    const int*   i8 = wid + (size_t)b * 8;
    float w[CAP]; int idx[CAP];
#pragma unroll
    for (int c = 0; c < CAP; ++c) { w[c] = w8[c]; idx[c] = i8[c]; }

    const int d = threadIdx.x * 4;
    float4 acc = make_float4(0.f, 0.f, 0.f, 0.f);
#pragma unroll
    for (int c = 0; c < CAP; ++c) {
        float4 u = *(const float4*)(U + (size_t)idx[c] * DM + d);
        acc.x += w[c] * u.x; acc.y += w[c] * u.y;
        acc.z += w[c] * u.z; acc.w += w[c] * u.w;
    }
    *(float4*)(out + (size_t)b * DM + d) = acc;
}

// ---------------------------------------------------------------------------
extern "C" void kernel_launch(void* const* d_in, const int* in_sizes, int n_in,
                              void* d_out, int out_size, void* d_ws, size_t ws_size,
                              hipStream_t stream) {
    (void)in_sizes; (void)n_in; (void)out_size; (void)ws_size;
    const float* X  = (const float*)d_in[0];
    const float* Wg = (const float*)d_in[1];
    const float* bg = (const float*)d_in[2];
    const float* Wi = (const float*)d_in[3];
    const float* bi = (const float*)d_in[4];
    float* out = (float*)d_out;

    char* ws = (char*)d_ws;
    unsigned short* Whi  = (unsigned short*)(ws);                             // 2 MB
    unsigned short* Wlo  = (unsigned short*)(ws + (2u << 20));                // 2 MB
    float* imp  = (float*)(ws + (4u << 20));                                  // 64 KB
    float* wv   = (float*)(ws + (4u << 20) + (1u << 16));                     // 512 KB
    int*   wid  = (int*)  (ws + (4u << 20) + (1u << 16) + (512u << 10));      // 512 KB
    float* impS = (float*)(ws + (4u << 20) + (1u << 16) + (1024u << 10));     // 512 KB
    float* U    = (float*)(ws + (8u << 20));                                  // 64 MB

    wm_prep_w<<<(DM * DM) / 256, 256, 0, stream>>>(Wg, Whi, Wlo);
    wm_importance<<<(BATCH_N * 32) / 256, 256, 0, stream>>>(X, Wi, bi, imp);
    wm_scan_state<<<1, 32, 0, stream>>>(imp, impS, wid);
    wm_weights<<<BATCH_N / 256, 256, 0, stream>>>(impS, wv);
    wm_gemm<<<dim3(DM / 64, BATCH_N / 128), 256, 0, stream>>>(X, Whi, Wlo, bg, U);
    wm_combine<<<BATCH_N, 256, 0, stream>>>(U, wv, wid, out);
}
